// NVFP4LinearLayer_5978594476493
// MI455X (gfx1250) — matrix operand order repaired and checked by reading, not compiled, not run on hardware
//
#include <hip/hip_runtime.h>

typedef _Float16 h2   __attribute__((ext_vector_type(2)));
typedef _Float16 v8h  __attribute__((ext_vector_type(8)));
typedef _Float16 v16h __attribute__((ext_vector_type(16)));
typedef float    v8f  __attribute__((ext_vector_type(8)));
typedef float    v4f  __attribute__((ext_vector_type(4)));

#define TM   256   // workgroup M tile
#define TN   128   // workgroup N tile
#define KT   32    // K step (one f16 WMMA K)
#define LDSW 40    // LDS row stride in f16 (32 data + 8 pad = 80B, bank-spread)

__device__ __forceinline__ unsigned cvt2(float a, float b) {
    h2 t = {(_Float16)a, (_Float16)b};
    return __builtin_bit_cast(unsigned, t);          // v_cvt_pk_f16_f32
}

__device__ __forceinline__ unsigned scmul(unsigned d, h2 sc) {
    h2 t = __builtin_bit_cast(h2, d) * sc;           // v_pk_mul_f16
    return __builtin_bit_cast(unsigned, t);
}

// Dequantize 4 packed-FP4 bytes (one per input int, lo nibble = even k) into
// 4 dwords of scaled f16 pairs. f16 magnitudes of {0,.5,1,1.5,2,3,4,6} have
// zero low bytes, so a single V_PERM byte-LUT per nibble-set suffices, and
// the fp4 sign bit ORs directly into the f16 high byte (bit7 = f16 sign).
__device__ __forceinline__ int4 dequant4(int4 wv, h2 sc) {
    unsigned p = __builtin_amdgcn_perm((unsigned)wv.y, (unsigned)wv.x, 0x0C0C0400u)
               | __builtin_amdgcn_perm((unsigned)wv.w, (unsigned)wv.z, 0x04000C0Cu);
    unsigned mL = __builtin_amdgcn_perm(0x46444240u, 0x3E3C3800u,  p        & 0x07070707u);
    unsigned mH = __builtin_amdgcn_perm(0x46444240u, 0x3E3C3800u, (p >> 4)  & 0x07070707u);
    mL |= (p & 0x08080808u) << 4;   // low-nibble signs -> byte bit7
    mH |=  p & 0x80808080u;         // high-nibble signs
    unsigned d0 = __builtin_amdgcn_perm(mH, mL, 0x040C000Cu); // {H0,0,L0,0}
    unsigned d1 = __builtin_amdgcn_perm(mH, mL, 0x050C010Cu);
    unsigned d2 = __builtin_amdgcn_perm(mH, mL, 0x060C020Cu);
    unsigned d3 = __builtin_amdgcn_perm(mH, mL, 0x070C030Cu);
    return make_int4((int)scmul(d0, sc), (int)scmul(d1, sc),
                     (int)scmul(d2, sc), (int)scmul(d3, sc));
}

__global__ __launch_bounds__(256)
void nvfp4_linear_wmma_f16(const float* __restrict__ x,
                           const int*   __restrict__ wdata,
                           const int*   __restrict__ wscale,
                           const float* __restrict__ gscale,
                           const float* __restrict__ bias,
                           float* __restrict__ out,
                           int M, int N, int K)
{
    __shared__ _Float16 xs[TM * LDSW];   // 20.0 KB
    __shared__ _Float16 ws[TN * LDSW];   // 10.0 KB

    const int tid  = threadIdx.x;
    const int lane = tid & 31;
    const int wave = tid >> 5;
    const int wm   = (wave >> 1) * 64;   // 4 waves along M
    const int wn   = (wave & 1) * 64;    // 2 waves along N

    const int nbase = blockIdx.x * TN;
    const int mbase = blockIdx.y * TM;
    const float gs  = gscale[0];

    // x loader: one row per thread, 32 k-values per step
    const float* xrow = x + (size_t)(mbase + tid) * K;
    // W loader: 2 threads per row, 16 k-values (= one scale group) each
    const int lr = tid >> 1;
    const int lh = tid & 1;
    const int*   wrow = wdata  + (size_t)(nbase + lr) * (K >> 1) + lh * 8;
    const int*   srow = wscale + (size_t)(nbase + lr) * (K >> 4) + lh;

    v8f acc[4][4];
    #pragma unroll
    for (int i = 0; i < 4; ++i)
        #pragma unroll
        for (int j = 0; j < 4; ++j)
            #pragma unroll
            for (int e = 0; e < 8; ++e)
                acc[i][j][e] = 0.0f;

    const int arow = lane & 15;
    const int ah   = lane >> 4;

    // ---------------- staged-register fetch of one K-step ----------------
    auto fetch = [&](int k0, int4 sx[4], int4 sw[2]) {
        v4f a0 = *(const v4f*)(xrow + k0 + 0);
        v4f a1 = *(const v4f*)(xrow + k0 + 4);
        v4f a2 = *(const v4f*)(xrow + k0 + 8);
        v4f a3 = *(const v4f*)(xrow + k0 + 12);
        v4f a4 = *(const v4f*)(xrow + k0 + 16);
        v4f a5 = *(const v4f*)(xrow + k0 + 20);
        v4f a6 = *(const v4f*)(xrow + k0 + 24);
        v4f a7 = *(const v4f*)(xrow + k0 + 28);
        int4 w0 = *(const int4*)(wrow + (k0 >> 1));
        int4 w1 = *(const int4*)(wrow + (k0 >> 1) + 4);
        int  sb = srow[k0 >> 4];

        // E4M3 group scale (sign bit never set in this data) * global scale
        unsigned e  = ((unsigned)sb >> 3) & 0xFu;
        unsigned mm = (unsigned)sb & 7u;
        float sv = __builtin_ldexpf((float)(e ? (mm + 8u) : mm),
                                    (int)(e ? e : 1u) - 10) * gs;
        _Float16 sh = (_Float16)sv;
        h2 sc = {sh, sh};

        sx[0] = make_int4((int)cvt2(a0.x,a0.y), (int)cvt2(a0.z,a0.w),
                          (int)cvt2(a1.x,a1.y), (int)cvt2(a1.z,a1.w));
        sx[1] = make_int4((int)cvt2(a2.x,a2.y), (int)cvt2(a2.z,a2.w),
                          (int)cvt2(a3.x,a3.y), (int)cvt2(a3.z,a3.w));
        sx[2] = make_int4((int)cvt2(a4.x,a4.y), (int)cvt2(a4.z,a4.w),
                          (int)cvt2(a5.x,a5.y), (int)cvt2(a5.z,a5.w));
        sx[3] = make_int4((int)cvt2(a6.x,a6.y), (int)cvt2(a6.z,a6.w),
                          (int)cvt2(a7.x,a7.y), (int)cvt2(a7.z,a7.w));
        sw[0] = dequant4(w0, sc);
        sw[1] = dequant4(w1, sc);
    };

    int4 sx[4], sw[2];
    fetch(0, sx, sw);

    for (int k0 = 0; k0 < K; k0 += KT) {
        __syncthreads();   // previous step's LDS readers done

        // serial region: pure LDS stores
        {
            _Float16* px = &xs[tid * LDSW];
            *(int4*)(px + 0)  = sx[0];
            *(int4*)(px + 8)  = sx[1];
            *(int4*)(px + 16) = sx[2];
            *(int4*)(px + 24) = sx[3];
            _Float16* pw = &ws[lr * LDSW + lh * 16];
            *(int4*)(pw + 0)  = sw[0];
            *(int4*)(pw + 8)  = sw[1];
        }
        __syncthreads();   // tile ready

        // prefetch next step (loads + conversion co-execute with WMMAs)
        const int kn = (k0 + KT < K) ? (k0 + KT) : k0;
        int4 nx[4], nw[2];
        fetch(kn, nx, nw);

        // fragments + 16 WMMAs
        v16h af[4], bf[4];
        #pragma unroll
        for (int mt = 0; mt < 4; ++mt) {
            // A 16x32: lanes 0-15 hold K {0..7,16..23}; lanes 16-31 offset +8
            const _Float16* pp = &xs[(wm + mt * 16 + arow) * LDSW + ah * 8];
            v8h lo = *(const v8h*)pp;
            v8h hi = *(const v8h*)(pp + 16);
            af[mt] = __builtin_shufflevector(lo, hi, 0,1,2,3,4,5,6,7,
                                                     8,9,10,11,12,13,14,15);
        }
        #pragma unroll
        for (int nt = 0; nt < 4; ++nt) {
            // B 32x16: lanes 0-15 hold K {0..7,16..23}; lanes 16-31 offset +8
            const _Float16* pp = &ws[(wn + nt * 16 + arow) * LDSW + ah * 8];
            v8h lo = *(const v8h*)pp;
            v8h hi = *(const v8h*)(pp + 16);
            bf[nt] = __builtin_shufflevector(lo, hi, 0,1,2,3,4,5,6,7,
                                                     8,9,10,11,12,13,14,15);
        }
        #pragma unroll
        for (int mt = 0; mt < 4; ++mt)
            #pragma unroll
            for (int nt = 0; nt < 4; ++nt)
                acc[mt][nt] = __builtin_amdgcn_wmma_f32_16x16x32_f16(
                    false, af[mt], false, bf[nt],
                    (short)0, acc[mt][nt], false, false);

        sx[0] = nx[0]; sx[1] = nx[1]; sx[2] = nx[2]; sx[3] = nx[3];
        sw[0] = nw[0]; sw[1] = nw[1];
    }

    // ---------------- epilogue: bias add + store ----------------
    #pragma unroll
    for (int nt = 0; nt < 4; ++nt) {
        const int n = nbase + wn + nt * 16 + arow;
        const float bv = bias[n];
        #pragma unroll
        for (int mt = 0; mt < 4; ++mt) {
            float* po = out + (size_t)(mbase + wm + mt * 16 + ah * 8) * N + n;
            #pragma unroll
            for (int v = 0; v < 8; ++v)
                po[(size_t)v * N] = acc[mt][nt][v] + bv;
        }
    }
}

extern "C" void kernel_launch(void* const* d_in, const int* in_sizes, int n_in,
                              void* d_out, int out_size, void* d_ws, size_t ws_size,
                              hipStream_t stream)
{
    (void)n_in; (void)out_size; (void)d_ws; (void)ws_size;
    const float* x      = (const float*)d_in[0];
    const int*   wdata  = (const int*)d_in[1];
    const int*   wscale = (const int*)d_in[2];
    const float* gs     = (const float*)d_in[3];
    const float* bias   = (const float*)d_in[4];
    float* out = (float*)d_out;

    const int N = in_sizes[4];              // OUT = 4096
    const int K = (in_sizes[2] / N) * 16;   // 4096
    const int M = in_sizes[0] / K;          // 8192

    dim3 grid(N / TN, M / TM);
    dim3 block(256);
    nvfp4_linear_wmma_f16<<<grid, block, 0, stream>>>(x, wdata, wscale, gs, bias,
                                                      out, M, N, K);
}
